// Model_33406255628589
// MI455X (gfx1250) — matrix operand order repaired and checked by reading, not compiled, not run on hardware
//
#include <hip/hip_runtime.h>
#include <hip/hip_bf16.h>

typedef __attribute__((ext_vector_type(16))) _Float16 v16h;
typedef __attribute__((ext_vector_type(8)))  _Float16 v8h;
typedef __attribute__((ext_vector_type(4)))  _Float16 v4h;
typedef __attribute__((ext_vector_type(8)))  float    v8f;

#define WMMA(A, B, C) \
  __builtin_amdgcn_wmma_f32_16x16x32_f16(false, (A), false, (B), (short)0, (C), false, false)

#define B_DIM   8
#define T_DIM   1024
#define HID     2048
#define NHEAD   16
#define HD      128
#define QRANK   128
#define KVRANK  64
#define ROWS    (B_DIM * T_DIM)          // 8192
#define MIN_VAL (-1e9f)
#define SCALE   0.08838834764831845f     // 1/sqrt(128)

__device__ __forceinline__ v8f zero8() {
  v8f r;
#pragma unroll
  for (int i = 0; i < 8; ++i) r[i] = 0.0f;
  return r;
}

// LDS byte offset of a __shared__ object: generic LDS pointers carry the LDS
// offset in addr[31:0] (CDNA5 aperture rules), so truncation is the conversion.
__device__ __forceinline__ unsigned lds_off(const void* p) {
  return (unsigned)(size_t)p;
}

// ---- A fragment (16x32 f16): M = lane%16 ; halves 0..15 -> K kb..kb+15 (kb = k0 + 16*(lane/16))
__device__ __forceinline__ v16h load_a_f16(const _Float16* __restrict__ A, int lda,
                                           int m0, int k0, int lane) {
  const _Float16* p = A + (size_t)(m0 + (lane & 15)) * lda + (k0 + ((lane & 16) ? 16 : 0));
  v16h r;
#pragma unroll
  for (int i = 0; i < 8; ++i) { r[i] = p[i]; r[i + 8] = p[8 + i]; }
  return r;
}

// A fragment out of an LDS-resident row-major f16 tile.
__device__ __forceinline__ v16h load_a_lds(const _Float16* base, int lda, int k0, int lane) {
  const _Float16* p = base + (lane & 15) * lda + k0 + ((lane & 16) ? 16 : 0);
  v16h r;
#pragma unroll
  for (int i = 0; i < 8; ++i) { r[i] = p[i]; r[i + 8] = p[8 + i]; }
  return r;
}

// ---- B fragment (32x16 f16) from [N][K]-major storage: N = lane%16, K contiguous.
__device__ __forceinline__ v16h load_b_nk(const _Float16* __restrict__ Bm, int ldb,
                                          int n0, int k0, int lane) {
  const _Float16* p = Bm + (size_t)(n0 + (lane & 15)) * ldb + (k0 + ((lane & 16) ? 16 : 0));
  v16h r;
#pragma unroll
  for (int i = 0; i < 8; ++i) { r[i] = p[i]; r[i + 8] = p[8 + i]; }
  return r;
}

// ---- B fragment from pre-packed tiles: tile = nt*KT + kt, 32 lanes x 16 halves contiguous.
__device__ __forceinline__ v16h load_b_packed(const _Float16* __restrict__ P,
                                              int nt, int kt, int KT, int lane) {
  const _Float16* p = P + (((size_t)(nt * KT + kt)) << 9) + lane * 16;
  v16h r;
#pragma unroll
  for (int i = 0; i < 8; ++i) { r[i] = p[i]; r[i + 8] = p[8 + i]; }
  return r;
}

// ================= pack: W[K][N] f32 row-major -> fragment-ordered f16 tiles ===============
__global__ void pack_b_tiles(const float* __restrict__ W, _Float16* __restrict__ P,
                             int K, int N) {
  int gid  = blockIdx.x * blockDim.x + threadIdx.x;
  int wave = gid >> 5, lane = gid & 31;
  int KT = K >> 5, NT = N >> 4;
  if (wave >= KT * NT) return;
  int nt = wave / KT, kt = wave % KT;
  int n  = nt * 16 + (lane & 15);
  int kb = kt * 32 + ((lane & 16) ? 16 : 0);
  const float* src = W + (size_t)kb * N + n;
  _Float16* dst = P + ((size_t)wave << 9) + lane * 16;
#pragma unroll
  for (int j = 0; j < 16; ++j) dst[j] = (_Float16)src[(size_t)j * N];
}

// ================= k1: h @ {Wq_a,Wkv_a} + bias, fused RMSNorm -> f16 activations ===========
// h tile (16 x 2048) is converted to f16 in LDS ONCE per block (two 32KB half-K phases),
// so the 12 GEMM waves read A fragments via ds_load instead of 12x-redundant cvt chains.
__global__ void __launch_bounds__(384)
k1_proj_a(const float* __restrict__ h, const _Float16* __restrict__ Pqa,
          const _Float16* __restrict__ Pkva, const float* __restrict__ bq_a,
          const float* __restrict__ bkv_a, const float* __restrict__ qnw,
          const float* __restrict__ kvnw, _Float16* __restrict__ Aq,
          _Float16* __restrict__ Akv) {
  __shared__ _Float16 hlds[16 * 1024];   // 32 KB staged half-K tile
  __shared__ float tile[16 * 192];       // 12 KB f32 output tile
  __shared__ float rq[16], rkv[16];
  const int tid = threadIdx.x, w = tid >> 5, lane = tid & 31;
  const int m0 = blockIdx.x * 16;
  const bool isq = (w < 8);
  const int nt = isq ? w : (w - 8);
  const _Float16* P = isq ? Pqa : Pkva;
  v8f acc = zero8();
  for (int ph = 0; ph < 2; ++ph) {
    // cooperative f32 -> f16 staging (vectorized 4-wide)
    for (int i = tid; i < 16 * 256; i += 384) {
      int row = i >> 8, c4 = (i & 255) * 4;
      const float4 f = *(const float4*)(h + (size_t)(m0 + row) * HID + ph * 1024 + c4);
      v4h hv;
      hv[0] = (_Float16)f.x; hv[1] = (_Float16)f.y;
      hv[2] = (_Float16)f.z; hv[3] = (_Float16)f.w;
      *(v4h*)(hlds + row * 1024 + c4) = hv;
    }
    __syncthreads();
#pragma unroll 4
    for (int kt = 0; kt < 32; ++kt) {
      v16h a = load_a_lds(hlds, 1024, kt * 32, lane);
      v16h b = load_b_packed(P, nt, ph * 32 + kt, 64, lane);
      acc = WMMA(a, b, acc);
    }
    __syncthreads();
  }
  const int nl   = nt * 16 + (lane & 15);
  const int col  = isq ? nl : (128 + nl);
  const float bs = isq ? bq_a[nl] : bkv_a[nl];
  const int roff = (lane & 16) ? 8 : 0;
#pragma unroll
  for (int r = 0; r < 8; ++r) tile[(r + roff) * 192 + col] = acc[r] + bs;
  __syncthreads();
  if (tid < 16) {
    float s = 0.f;
    for (int c = 0; c < 128; ++c) { float v = tile[tid * 192 + c]; s += v * v; }
    rq[tid] = rsqrtf(s * (1.0f / 128.0f) + 1e-6f);
  } else if (tid >= 32 && tid < 48) {
    int row = tid - 32; float s = 0.f;
    for (int c = 128; c < 192; ++c) { float v = tile[row * 192 + c]; s += v * v; }
    rkv[row] = rsqrtf(s * (1.0f / 64.0f) + 1e-6f);
  }
  __syncthreads();
  for (int i = tid; i < 16 * 128; i += 384) {
    int row = i >> 7, c = i & 127;
    Aq[(size_t)(m0 + row) * QRANK + c] = (_Float16)(tile[row * 192 + c] * rq[row] * qnw[c]);
  }
  for (int i = tid; i < 16 * 64; i += 384) {
    int row = i >> 6, c = i & 63;
    Akv[(size_t)(m0 + row) * KVRANK + c] = (_Float16)(tile[row * 192 + 128 + c] * rkv[row] * kvnw[c]);
  }
}

// ================= k2: LoRA-b GEMMs + bias -> q_raw [8192,2048], kv_raw [8192,256] =========
__global__ void __launch_bounds__(256)
k2_proj_b(const _Float16* __restrict__ Aq, const _Float16* __restrict__ Akv,
          const _Float16* __restrict__ Pqb, const _Float16* __restrict__ Pkvb,
          const float* __restrict__ bq_b, const float* __restrict__ bkv_b,
          _Float16* __restrict__ q_raw, _Float16* __restrict__ kv_raw) {
  const int tid = threadIdx.x, w = tid >> 5, lane = tid & 31;
  const int m0 = blockIdx.x * 16;
  const int roff = (lane & 16) ? 8 : 0;
  v16h aq[4], akv[2];
#pragma unroll
  for (int kc = 0; kc < 4; ++kc) aq[kc] = load_a_f16(Aq, QRANK, m0, kc * 32, lane);
#pragma unroll
  for (int kc = 0; kc < 2; ++kc) akv[kc] = load_a_f16(Akv, KVRANK, m0, kc * 32, lane);
  for (int nt = w; nt < 144; nt += 8) {
    v8f c = zero8();
    if (nt < 128) {
#pragma unroll
      for (int kc = 0; kc < 4; ++kc) c = WMMA(aq[kc], load_b_packed(Pqb, nt, kc, 4, lane), c);
      int n = nt * 16 + (lane & 15); float bs = bq_b[n];
#pragma unroll
      for (int r = 0; r < 8; ++r)
        q_raw[(size_t)(m0 + r + roff) * 2048 + n] = (_Float16)(c[r] + bs);
    } else {
      int t2 = nt - 128;
#pragma unroll
      for (int kc = 0; kc < 2; ++kc) c = WMMA(akv[kc], load_b_packed(Pkvb, t2, kc, 2, lane), c);
      int n = t2 * 16 + (lane & 15); float bs = bkv_b[n];
#pragma unroll
      for (int r = 0; r < 8; ++r)
        kv_raw[(size_t)(m0 + r + roff) * 256 + n] = (_Float16)(c[r] + bs);
    }
  }
}

// ================= k3: blended RoPE + nope mix -> q_f, k_f (row-major), v_t (d-major) ======
__global__ void __launch_bounds__(128)
k3_rope(const _Float16* __restrict__ q_raw, const _Float16* __restrict__ kv_raw,
        const float* __restrict__ cg, const float* __restrict__ sg,
        const float* __restrict__ cl, const float* __restrict__ sl,
        const float* __restrict__ nope_logit, const float* __restrict__ gl_logit,
        _Float16* __restrict__ q_f, _Float16* __restrict__ k_f, _Float16* __restrict__ v_t) {
  const int bt = blockIdx.x, b = bt >> 10, t = bt & 1023;
  const float a  = 1.0f / (1.0f + __expf(-gl_logit[0]));
  const float na = 1.0f / (1.0f + __expf(-nope_logit[0]));
  const _Float16* qr = q_raw + (size_t)bt * 2048;
  _Float16* qf = q_f + (size_t)bt * 2048;
  for (int i = threadIdx.x; i < 2048; i += 128) {
    int d = i & 127, dd = d & 63;
    float c = cg[t * 64 + dd] * a + cl[t * 64 + dd] * (1.0f - a);
    float s = sg[t * 64 + dd] * a + sl[t * 64 + dd] * (1.0f - a);
    float x  = (float)qr[i];
    float xp = (float)qr[(d < 64) ? i + 64 : i - 64];
    float pe = (d < 64) ? (x * c - xp * s) : (x * c + xp * s);
    qf[i] = (_Float16)(na * x + (1.0f - na) * pe);
  }
  const _Float16* kv = kv_raw + (size_t)bt * 256;
  for (int i = threadIdx.x; i < 128; i += 128) {
    int dd = i & 63;
    float c = cg[t * 64 + dd] * a + cl[t * 64 + dd] * (1.0f - a);
    float s = sg[t * 64 + dd] * a + sl[t * 64 + dd] * (1.0f - a);
    float x  = (float)kv[i];
    float xp = (float)kv[(i < 64) ? i + 64 : i - 64];
    float pe = (i < 64) ? (x * c - xp * s) : (x * c + xp * s);
    k_f[(size_t)bt * HD + i] = (_Float16)(na * x + (1.0f - na) * pe);
    v_t[(size_t)b * HD * T_DIM + (size_t)i * T_DIM + t] = kv[128 + i];
  }
}

// ================= k4: flash attention, one wave per (b, h, 16-query tile) =================
__global__ void __launch_bounds__(256)
k4_attn(const _Float16* __restrict__ q_f, const _Float16* __restrict__ k_f,
        const _Float16* __restrict__ v_t, const int* __restrict__ amask,
        _Float16* __restrict__ attn_out) {
  __shared__ _Float16 plds[8][16 * 32];
  const int tid = threadIdx.x, w = tid >> 5, lane = tid & 31;
  const int wg = blockIdx.x * 8 + w;
  const int b = wg >> 10, rem = wg & 1023, hh = rem >> 6, qt = rem & 63;
  const int q0 = qt * 16;
  const int roff = (lane & 16) ? 8 : 0;
  const _Float16* qfb = q_f + (size_t)b * T_DIM * 2048 + hh * HD;
  const _Float16* kfb = k_f + (size_t)b * T_DIM * HD;
  const _Float16* vtb = v_t + (size_t)b * HD * T_DIM;
  const int* amb = amask + b * T_DIM;
  _Float16* mylds = &plds[w][0];

  v16h aq[4];
#pragma unroll
  for (int kc = 0; kc < 4; ++kc) aq[kc] = load_a_f16(qfb, 2048, q0, kc * 32, lane);

  v8f o[8];
#pragma unroll
  for (int nt = 0; nt < 8; ++nt) o[nt] = zero8();
  float m[8], l[8];
#pragma unroll
  for (int r = 0; r < 8; ++r) { m[r] = -1e30f; l[r] = 0.0f; }

  const int nk = q0 + 16;  // number of causal keys
  for (int kb = 0; kb < nk; kb += 32) {
    v8f s0 = zero8(), s1 = zero8();
#pragma unroll
    for (int kc = 0; kc < 4; ++kc) {
      s0 = WMMA(aq[kc], load_b_nk(kfb, HD, kb, kc * 32, lane), s0);
      s1 = WMMA(aq[kc], load_b_nk(kfb, HD, kb + 16, kc * 32, lane), s1);
    }
    const int key0 = kb + (lane & 15), key1 = key0 + 16;
    const float pm0 = amb[key0] ? 0.0f : MIN_VAL;
    const float pm1 = amb[key1] ? 0.0f : MIN_VAL;
    float al[8];
#pragma unroll
    for (int r = 0; r < 8; ++r) {
      const int qi = q0 + r + roff;
      float x0 = s0[r] * SCALE + ((key0 > qi) ? MIN_VAL : pm0);
      float x1 = s1[r] * SCALE + ((key1 > qi) ? MIN_VAL : pm1);
      float mx = fmaxf(x0, x1);
#pragma unroll
      for (int off = 1; off < 16; off <<= 1) mx = fmaxf(mx, __shfl_xor(mx, off, 32));
      float mnew  = fmaxf(m[r], mx);
      float alpha = __expf(m[r] - mnew);
      float p0 = __expf(x0 - mnew), p1 = __expf(x1 - mnew);
      float rs = p0 + p1;
#pragma unroll
      for (int off = 1; off < 16; off <<= 1) rs += __shfl_xor(rs, off, 32);
      l[r] = l[r] * alpha + rs;
      m[r] = mnew;
      al[r] = alpha;
      mylds[(r + roff) * 32 + (lane & 15)]      = (_Float16)p0;
      mylds[(r + roff) * 32 + 16 + (lane & 15)] = (_Float16)p1;
    }
#pragma unroll
    for (int nt = 0; nt < 8; ++nt)
#pragma unroll
      for (int r = 0; r < 8; ++r) o[nt][r] *= al[r];
    // P tile back out of wave-private LDS as an A fragment (lda = 32)
    v16h pa;
    {
      const _Float16* p = mylds + (lane & 15) * 32 + ((lane & 16) ? 16 : 0);
#pragma unroll
      for (int i = 0; i < 8; ++i) { pa[i] = p[i]; pa[i + 8] = p[8 + i]; }
    }
#pragma unroll
    for (int nt = 0; nt < 8; ++nt)
      o[nt] = WMMA(pa, load_b_nk(vtb, T_DIM, nt * 16, kb, lane), o[nt]);
  }
#pragma unroll
  for (int nt = 0; nt < 8; ++nt) {
#pragma unroll
    for (int r = 0; r < 8; ++r) {
      size_t row = (size_t)(b * T_DIM + q0 + r + roff);
      attn_out[row * 2048 + hh * HD + nt * 16 + (lane & 15)] = (_Float16)(o[nt][r] / l[r]);
    }
  }
}

// ================= k5: attn_out [8192,2048] @ Wo -> f32 ====================================
// Packed Wo tiles are streamed into LDS with GLOBAL_LOAD_ASYNC_TO_LDS_B128 (ASYNCcnt),
// double-buffered so tile kt+1 is in flight while tile kt feeds the 4 WMMAs.
__global__ void __launch_bounds__(256)
k5_wo(const _Float16* __restrict__ attn, const _Float16* __restrict__ PWo,
      float* __restrict__ out) {
  __shared__ _Float16 btile[8][2][512];  // per-wave double buffer, 1KB each
  const int tid = threadIdx.x, w = tid >> 5, lane = tid & 31;
  const int nt = blockIdx.y * 8 + w;
  const int m0 = blockIdx.x * 64;
  const int roff = (lane & 16) ? 8 : 0;
  const unsigned lbase = lds_off(&btile[w][0][0]);

  auto issue = [&](int kt, int buf) {
    const char* g = (const char*)(PWo + (((size_t)(nt * 64 + kt)) << 9));
    unsigned long long g0 = (unsigned long long)(g + lane * 16);
    unsigned long long g1 = (unsigned long long)(g + 512 + lane * 16);
    unsigned l0 = lbase + (unsigned)buf * 1024u + lane * 16;
    unsigned l1 = l0 + 512u;
    asm volatile("global_load_async_to_lds_b128 %0, %1, off" :: "v"(l0), "v"(g0) : "memory");
    asm volatile("global_load_async_to_lds_b128 %0, %1, off" :: "v"(l1), "v"(g1) : "memory");
  };

  v8f acc[4];
#pragma unroll
  for (int i = 0; i < 4; ++i) acc[i] = zero8();

  issue(0, 0);
  for (int kt = 0; kt < 64; ++kt) {
    if (kt + 1 < 64) {
      issue(kt + 1, (kt + 1) & 1);
      asm volatile("s_wait_asynccnt 0x2" ::: "memory");  // current tile landed
    } else {
      asm volatile("s_wait_asynccnt 0x0" ::: "memory");
    }
    const _Float16* p = &btile[w][kt & 1][lane * 16];
    v16h b;
#pragma unroll
    for (int i = 0; i < 8; ++i) { b[i] = p[i]; b[i + 8] = p[8 + i]; }
#pragma unroll
    for (int i = 0; i < 4; ++i)
      acc[i] = WMMA(load_a_f16(attn, 2048, m0 + i * 16, kt * 32, lane), b, acc[i]);
  }
#pragma unroll
  for (int i = 0; i < 4; ++i)
#pragma unroll
    for (int r = 0; r < 8; ++r)
      out[(size_t)(m0 + i * 16 + r + roff) * 2048 + nt * 16 + (lane & 15)] = acc[i][r];
}

// =========================================================================================
extern "C" void kernel_launch(void* const* d_in, const int* in_sizes, int n_in,
                              void* d_out, int out_size, void* d_ws, size_t ws_size,
                              hipStream_t stream) {
  (void)in_sizes; (void)n_in; (void)out_size; (void)ws_size;
  const float* h      = (const float*)d_in[0];
  const float* cos_g  = (const float*)d_in[1];
  const float* sin_g  = (const float*)d_in[2];
  const float* cos_l  = (const float*)d_in[3];
  const float* sin_l  = (const float*)d_in[4];
  const float* Wq_a   = (const float*)d_in[5];
  const float* bq_a   = (const float*)d_in[6];
  const float* Wq_b   = (const float*)d_in[7];
  const float* bq_b   = (const float*)d_in[8];
  const float* Wkv_a  = (const float*)d_in[9];
  const float* bkv_a  = (const float*)d_in[10];
  const float* Wkv_b  = (const float*)d_in[11];
  const float* bkv_b  = (const float*)d_in[12];
  const float* qnw    = (const float*)d_in[13];
  const float* kvnw   = (const float*)d_in[14];
  const float* Wo     = (const float*)d_in[15];
  const float* nopel  = (const float*)d_in[16];
  const float* gll    = (const float*)d_in[17];
  const int*   amask  = (const int*)d_in[18];
  float* out = (float*)d_out;

  char* ws = (char*)d_ws;
  auto carve = [&](size_t bytes) { char* p = ws; ws += (bytes + 255) & ~(size_t)255; return p; };
  _Float16* Pqa    = (_Float16*)carve((size_t)HID * QRANK * 2);
  _Float16* Pkva   = (_Float16*)carve((size_t)HID * KVRANK * 2);
  _Float16* Pqb    = (_Float16*)carve((size_t)QRANK * (NHEAD * HD) * 2);
  _Float16* Pkvb   = (_Float16*)carve((size_t)KVRANK * 256 * 2);
  _Float16* PWo    = (_Float16*)carve((size_t)(NHEAD * HD) * HID * 2);
  _Float16* Aq     = (_Float16*)carve((size_t)ROWS * QRANK * 2);
  _Float16* Akv    = (_Float16*)carve((size_t)ROWS * KVRANK * 2);
  _Float16* q_raw  = (_Float16*)carve((size_t)ROWS * 2048 * 2);
  _Float16* kv_raw = (_Float16*)carve((size_t)ROWS * 256 * 2);
  _Float16* q_f    = (_Float16*)carve((size_t)ROWS * 2048 * 2);
  _Float16* k_f    = (_Float16*)carve((size_t)ROWS * HD * 2);
  _Float16* v_t    = (_Float16*)carve((size_t)ROWS * HD * 2);
  _Float16* attn   = q_raw;  // q_raw dead after k3; reuse for attention output

  auto pack = [&](const float* W, _Float16* P, int K, int N) {
    int waves = (K >> 5) * (N >> 4);
    int thr = waves * 32;
    pack_b_tiles<<<(thr + 255) / 256, 256, 0, stream>>>(W, P, K, N);
  };
  pack(Wq_a, Pqa, HID, QRANK);
  pack(Wkv_a, Pkva, HID, KVRANK);
  pack(Wq_b, Pqb, QRANK, NHEAD * HD);
  pack(Wkv_b, Pkvb, KVRANK, 256);
  pack(Wo, PWo, NHEAD * HD, HID);

  k1_proj_a<<<ROWS / 16, 384, 0, stream>>>(h, Pqa, Pkva, bq_a, bkv_a, qnw, kvnw, Aq, Akv);
  k2_proj_b<<<ROWS / 16, 256, 0, stream>>>(Aq, Akv, Pqb, Pkvb, bq_b, bkv_b, q_raw, kv_raw);
  k3_rope<<<ROWS, 128, 0, stream>>>(q_raw, kv_raw, cos_g, sin_g, cos_l, sin_l,
                                    nopel, gll, q_f, k_f, v_t);
  k4_attn<<<(B_DIM * NHEAD * (T_DIM / 16)) / 8, 256, 0, stream>>>(q_f, k_f, v_t, amask, attn);
  k5_wo<<<dim3(ROWS / 64, 16), 256, 0, stream>>>(attn, PWo, out);
}
